// HOIST_61864708932289
// MI455X (gfx1250) — matrix operand order repaired and checked by reading, not compiled, not run on hardware
//
#include <hip/hip_runtime.h>

// ---------- types ----------
typedef _Float16 h8  __attribute__((ext_vector_type(8)));
typedef _Float16 h16 __attribute__((ext_vector_type(16)));
typedef float    f8v __attribute__((ext_vector_type(8)));

#define NNODE 1024
#define TT    128
#define DD    128
#define HH    128
#define FF    144
#define SD    16
#define OO    32
#define NT    (NNODE*TT)

__device__ __forceinline__ float sigmoidf_(float x){ return 1.0f/(1.0f + __expf(-x)); }

// ---------- WMMA fragment loaders (f16, 16x16x32) ----------
// A: 16(M) x 32(K).  lane L: M = L&15 ; halves 0..7 = K k0+8*(L>>4&1) .. +7 ; halves 8..15 = +16
__device__ __forceinline__ h16 load_a_frag(const _Float16* base, int ld, int row0, int k0){
    int L = threadIdx.x & 31;
    const _Float16* p = base + (size_t)(row0 + (L & 15)) * ld + k0 + ((L >> 4) & 1) * 8;
    h16 a;
    *(h8*)&a        = *(const h8*)p;
    *((h8*)&a + 1)  = *(const h8*)(p + 16);
    return a;
}
// B: 32(K) x 16(Ncols), stored N-major: B[col][k], row stride ld.
// lane L: col = col0 + (L&15); halves 0..7 = K k0+8*(L>>4&1) .. +7 ; halves 8..15 = +16
__device__ __forceinline__ h16 load_b_frag(const _Float16* base, int ld, int col0, int k0){
    int L = threadIdx.x & 31;
    const _Float16* p = base + (size_t)(col0 + (L & 15)) * ld + k0 + ((L >> 4) & 1) * 8;
    h16 b;
    *(h8*)&b       = *(const h8*)p;
    *((h8*)&b + 1) = *(const h8*)(p + 16);
    return b;
}
#define WMMA_F16(a,b,c) __builtin_amdgcn_wmma_f32_16x16x32_f16(false,(a),false,(b),(short)0,(c),false,false)

// ============================================================
// Kernel 1: u[m], v[m] (rank-1 GAT score) and wa = W_dis @ a_dis
// ============================================================
__global__ void prep_kernel(const float* __restrict__ x, const float* __restrict__ w_stat,
                            const float* __restrict__ a_stat, const float* __restrict__ W_dis,
                            const float* __restrict__ a_dis,
                            float* __restrict__ u, float* __restrict__ v, float* __restrict__ wa){
    int tid = blockIdx.x * 256 + threadIdx.x;
    if (tid < 8){
        float s = 0.f;
        #pragma unroll
        for (int j = 0; j < 8; j++) s += W_dis[tid*8 + j] * a_dis[j];
        wa[tid] = s;
    }
    if (tid < NNODE){
        float st[SD];
        const float* xp = x + (size_t)tid * TT * FF;   // t = 0 slice
        #pragma unroll
        for (int k = 0; k < SD; k++) st[k] = xp[k];
        float uu = 0.f, vv = 0.f;
        for (int i = 0; i < SD; i++){
            float h = 0.f;
            #pragma unroll
            for (int k = 0; k < SD; k++) h += st[k] * w_stat[k*SD + i];
            uu += h * a_stat[i];
            vv += h * a_stat[SD + i];
        }
        u[tid] = uu; v[tid] = vv;
    }
}

// ============================================================
// Kernel 2: convert weights to f16 N-major layouts
// ============================================================
__global__ void convw_kernel(const float* __restrict__ w1, const float* __restrict__ w2,
                             const float* __restrict__ wih, const float* __restrict__ whh,
                             const float* __restrict__ linw,
                             _Float16* w1h, _Float16* w2h, _Float16* wihh, _Float16* whhh,
                             _Float16* linwh){
    int idx = blockIdx.x * 256 + threadIdx.x;
    if (idx < 16384){                 // mlp_w1 (DD,H) -> [j][k]
        int j = idx >> 7, k = idx & 127; w1h[idx] = (_Float16)w1[k*HH + j];
    } else if (idx < 32768){          // mlp_w2 (H,DD) -> [j][k]
        int i = idx - 16384; int j = i >> 7, k = i & 127; w2h[i] = (_Float16)w2[k*DD + j];
    } else if (idx < 98304){          // W_ih (4H,DD): already [j][k]
        int i = idx - 32768; wihh[i] = (_Float16)wih[i];
    } else if (idx < 163840){         // W_hh (4H,H): already [j][k]
        int i = idx - 98304; whhh[i] = (_Float16)whh[i];
    } else if (idx < 180224){         // lin_w (H,H) -> [j][k]
        int i = idx - 163840; int j = i >> 7, k = i & 127; linwh[i] = (_Float16)linw[k*HH + j];
    }
}

// ============================================================
// Kernel 3: fused pairwise scores + row softmax -> A (f16)
// ============================================================
__global__ void attn_kernel(const float* __restrict__ dist, const float* __restrict__ u,
                            const float* __restrict__ v, const float* __restrict__ wa,
                            _Float16* __restrict__ Ah){
    __shared__ float sc[NNODE];
    __shared__ float red[256];
    __shared__ float swa[8];
    int m = blockIdx.x, tid = threadIdx.x;
    if (tid < 8) swa[tid] = wa[tid];
    __syncthreads();
    float um = u[m];
    float lmax = -1e30f;
    for (int n = tid; n < NNODE; n += 256){
        const float* dp = dist + ((size_t)m * NNODE + n) * 8;
        float dd = 0.f;
        #pragma unroll
        for (int k = 0; k < 8; k++) dd += dp[k] * swa[k];
        float s = sigmoidf_(um + v[n]) + sigmoidf_(dd);
        sc[n] = s;
        lmax = fmaxf(lmax, s);
    }
    red[tid] = lmax; __syncthreads();
    for (int off = 128; off > 0; off >>= 1){
        if (tid < off) red[tid] = fmaxf(red[tid], red[tid + off]);
        __syncthreads();
    }
    float mx = red[0]; __syncthreads();
    float lsum = 0.f;
    for (int n = tid; n < NNODE; n += 256){
        float e = __expf(sc[n] - mx); sc[n] = e; lsum += e;
    }
    red[tid] = lsum; __syncthreads();
    for (int off = 128; off > 0; off >>= 1){
        if (tid < off) red[tid] = red[tid] + red[tid + off];
        __syncthreads();
    }
    float inv = 1.0f / red[0];
    for (int n = tid; n < NNODE; n += 256)
        Ah[(size_t)m * NNODE + n] = (_Float16)(sc[n] * inv);
}

// ============================================================
// Kernel 4: gating MLP -> lstm_in (f16 [n][t][d]); 2 WMMA GEMMs
// block = 16 rows of (n,t), 8 waves, each wave one 16x16 col tile
// ============================================================
__global__ void gate_kernel(const float* __restrict__ x, const _Float16* __restrict__ w1h,
                            const _Float16* __restrict__ w2h, const float* __restrict__ b1,
                            const float* __restrict__ b2, _Float16* __restrict__ lstmin){
    __shared__ _Float16 dynh[16*128];
    __shared__ _Float16 tmph[16*128];
    int tid = threadIdx.x;
    int row0 = blockIdx.x * 16;                    // global row (n*T + t)
    {
        int r = tid >> 4, c8 = (tid & 15) * 8;
        const float* xp = x + (size_t)(row0 + r) * FF + SD + c8;
        #pragma unroll
        for (int i = 0; i < 8; i++) dynh[r*128 + c8 + i] = (_Float16)xp[i];
    }
    __syncthreads();
    int w = tid >> 5, L = tid & 31;
    int col0 = 16 * w;
    f8v acc;
    {
        float bi = b1[col0 + (L & 15)];
        #pragma unroll
        for (int i = 0; i < 8; i++) acc[i] = bi;
    }
    #pragma unroll
    for (int kc = 0; kc < 4; kc++){
        h16 a = load_a_frag(dynh, 128, 0, kc*32);
        h16 b = load_b_frag(w1h, 128, col0, kc*32);
        acc = WMMA_F16(a, b, acc);
    }
    #pragma unroll
    for (int r = 0; r < 8; r++){
        float vv = acc[r];
        vv = vv > 0.f ? vv : 0.01f * vv;           // leaky_relu
        int rr = r + 8 * ((L >> 4) & 1);
        tmph[rr*128 + col0 + (L & 15)] = (_Float16)vv;
    }
    __syncthreads();
    {
        float bi = b2[col0 + (L & 15)];
        #pragma unroll
        for (int i = 0; i < 8; i++) acc[i] = bi;
    }
    #pragma unroll
    for (int kc = 0; kc < 4; kc++){
        h16 a = load_a_frag(tmph, 128, 0, kc*32);
        h16 b = load_b_frag(w2h, 128, col0, kc*32);
        acc = WMMA_F16(a, b, acc);
    }
    #pragma unroll
    for (int r = 0; r < 8; r++){
        int rr = r + 8 * ((L >> 4) & 1);
        int d  = col0 + (L & 15);
        float g  = sigmoidf_(acc[r]);
        float dv = (float)dynh[rr*128 + d];
        lstmin[(size_t)(row0 + rr) * DD + d] = (_Float16)(g * dv);
    }
}

// ============================================================
// Kernel 5: LSTM over T. Block = 16 nodes, 8 waves.
// Wave w owns hidden slice d in [16w,16w+16): i/f/g/o tiles align with
// the WMMA C layout so the c-state lives entirely in registers (v8f).
// Writes hs f32 [n][t][d] and hs^T f16 [t*H+d][n].
// ============================================================
__global__ void lstm_kernel(const _Float16* __restrict__ lstmin, const _Float16* __restrict__ wih,
                            const _Float16* __restrict__ whh, const float* __restrict__ bih,
                            const float* __restrict__ bhh,
                            float* __restrict__ hsf_g, _Float16* __restrict__ hst){
    __shared__ _Float16 xh[16*128];
    __shared__ _Float16 hh[16*128];
    __shared__ float    hf[16*128];
    int tid = threadIdx.x, w = tid >> 5, L = tid & 31;
    int n0 = blockIdx.x * 16;
    for (int i = tid; i < 16*128; i += 256) hh[i] = (_Float16)0.f;
    int d = 16*w + (L & 15);
    float bsum[4];
    #pragma unroll
    for (int g = 0; g < 4; g++){ int col = g*HH + d; bsum[g] = bih[col] + bhh[col]; }
    f8v c;
    #pragma unroll
    for (int i = 0; i < 8; i++) c[i] = 0.f;

    for (int t = 0; t < TT; t++){
        {   // stage x_t tile (coalesced) + prefetch next step's slab
            int r = tid >> 4, c8 = (tid & 15) * 8;
            const _Float16* xp = lstmin + ((size_t)(n0 + r) * TT + t) * DD + c8;
            *(h8*)(xh + r*128 + c8) = *(const h8*)xp;
            if (t + 1 < TT)
                __builtin_prefetch(lstmin + ((size_t)(n0 + r) * TT + (t + 1)) * DD + c8, 0, 3);
        }
        __syncthreads();
        f8v gac[4];
        #pragma unroll
        for (int g = 0; g < 4; g++){
            f8v a;
            #pragma unroll
            for (int i = 0; i < 8; i++) a[i] = bsum[g];
            int col0 = g*HH + 16*w;
            #pragma unroll
            for (int kc = 0; kc < 4; kc++){
                h16 af = load_a_frag(xh, 128, 0, kc*32);
                h16 bf = load_b_frag(wih, 128, col0, kc*32);
                a = WMMA_F16(af, bf, a);
            }
            #pragma unroll
            for (int kc = 0; kc < 4; kc++){
                h16 af = load_a_frag(hh, 128, 0, kc*32);
                h16 bf = load_b_frag(whh, 128, col0, kc*32);
                a = WMMA_F16(af, bf, a);
            }
            gac[g] = a;
        }
        __syncthreads();            // all waves done reading hh
        #pragma unroll
        for (int r = 0; r < 8; r++){
            float iv = sigmoidf_(gac[0][r]);
            float fv = sigmoidf_(gac[1][r]);
            float gv = tanhf(gac[2][r]);
            float ov = sigmoidf_(gac[3][r]);
            float cv = fv * c[r] + iv * gv;
            c[r] = cv;
            float hv = ov * tanhf(cv);
            int rr = r + 8 * ((L >> 4) & 1);
            hh[rr*128 + d] = (_Float16)hv;
            hf[rr*128 + d] = hv;
        }
        __syncthreads();
        {   // coalesced hs writes
            int r = tid >> 4, c8 = (tid & 15) * 8;
            float* op = hsf_g + ((size_t)(n0 + r) * TT + t) * HH + c8;
            #pragma unroll
            for (int i = 0; i < 8; i++) op[i] = hf[r*128 + c8 + i];
            if (tid < 128){
                _Float16 tmp[16];
                #pragma unroll
                for (int i = 0; i < 16; i++) tmp[i] = hh[i*128 + tid];
                _Float16* tp = hst + ((size_t)t * HH + tid) * NNODE + n0;
                #pragma unroll
                for (int i = 0; i < 16; i++) tp[i] = tmp[i];
            }
        }
        __syncthreads();
    }
}

// ============================================================
// Kernel 6: mix = hs + A@hs (K=1024 WMMA), then y1 = mix@lin_w + b,
// then s = y1@lin1_w + b -> sbuf[n][t].
// Block = 16 nodes x TWO t-columns: one A-fragment feeds two WMMAs
// (halves A traffic, 2 independent accumulator chains for ILP).
// ============================================================
__global__ void mix_kernel(const _Float16* __restrict__ Ah, const _Float16* __restrict__ hst,
                           const float* __restrict__ hsf, const _Float16* __restrict__ linwh,
                           const float* __restrict__ linb, const float* __restrict__ lin1w,
                           const float* __restrict__ lin1b, float* __restrict__ sbuf){
    __shared__ _Float16 mixh[16*128];
    __shared__ float    y1s[16*128];
    __shared__ float    ps[16*16];
    int tid = threadIdx.x, w = tid >> 5, L = tid & 31;
    int m0 = blockIdx.x * 16;
    int t0 = blockIdx.y * 2;
    int col0 = 16 * w;
    f8v acc0, acc1;
    #pragma unroll
    for (int i = 0; i < 8; i++){ acc0[i] = 0.f; acc1[i] = 0.f; }
    for (int k = 0; k < NNODE; k += 32){
        h16 a  = load_a_frag(Ah, NNODE, m0, k);
        h16 b0 = load_b_frag(hst, NNODE, (t0    )*HH + col0, k);
        h16 b1 = load_b_frag(hst, NNODE, (t0 + 1)*HH + col0, k);
        acc0 = WMMA_F16(a, b0, acc0);
        acc1 = WMMA_F16(a, b1, acc1);
        if (k + 64 < NNODE){
            __builtin_prefetch(Ah  + (size_t)(m0 + (L & 15)) * NNODE + k + 64, 0, 3);
            __builtin_prefetch(hst + (size_t)((t0    )*HH + col0 + (L & 15)) * NNODE + k + 64, 0, 3);
            __builtin_prefetch(hst + (size_t)((t0 + 1)*HH + col0 + (L & 15)) * NNODE + k + 64, 0, 3);
        }
    }
    #pragma unroll
    for (int tt = 0; tt < 2; tt++){
        int t = t0 + tt;
        __syncthreads();            // previous iteration fully done with mixh/y1s/ps
        #pragma unroll
        for (int r = 0; r < 8; r++){
            int rr = r + 8 * ((L >> 4) & 1);
            int dd = col0 + (L & 15);
            float av = (tt == 0) ? acc0[r] : acc1[r];
            float vv = av + hsf[((size_t)(m0 + rr) * TT + t) * HH + dd];  // residual
            mixh[rr*128 + dd] = (_Float16)vv;
        }
        __syncthreads();
        f8v a2;
        {
            float bi = linb[col0 + (L & 15)];
            #pragma unroll
            for (int i = 0; i < 8; i++) a2[i] = bi;
        }
        #pragma unroll
        for (int kc = 0; kc < 4; kc++){
            h16 a = load_a_frag(mixh, 128, 0, kc*32);
            h16 b = load_b_frag(linwh, 128, col0, kc*32);
            a2 = WMMA_F16(a, b, a2);
        }
        #pragma unroll
        for (int r = 0; r < 8; r++){
            int rr = r + 8 * ((L >> 4) & 1);
            y1s[rr*128 + col0 + (L & 15)] = a2[r];
        }
        __syncthreads();
        {
            int r = tid >> 4, p = tid & 15;
            float part = 0.f;
            #pragma unroll
            for (int j = 0; j < 8; j++){ int jj = p*8 + j; part += y1s[r*128 + jj] * lin1w[jj]; }
            ps[r*16 + p] = part;
        }
        __syncthreads();
        if (tid < 16){
            float s = lin1b[0];
            #pragma unroll
            for (int p = 0; p < 16; p++) s += ps[tid*16 + p];
            sbuf[(size_t)(m0 + tid) * TT + t] = s;
        }
    }
}

// ============================================================
// Kernel 7: y = leaky_relu(s) @ lin2_w + lin2_b  -> out (8,128,32)
// ============================================================
__global__ void final_kernel(const float* __restrict__ sbuf, const float* __restrict__ lin2w,
                             const float* __restrict__ lin2b, float* __restrict__ out){
    int idx = blockIdx.x * 256 + threadIdx.x;     // 32768
    int n = idx >> 5, o = idx & 31;
    float acc = lin2b[o];
    const float* sp = sbuf + (size_t)n * TT;
    for (int t = 0; t < TT; t++){
        float sv = sp[t];
        sv = sv > 0.f ? sv : 0.01f * sv;
        acc += sv * lin2w[t*OO + o];
    }
    out[idx] = acc;
}

// ============================================================
// host launcher
// ============================================================
extern "C" void kernel_launch(void* const* d_in, const int* in_sizes, int n_in,
                              void* d_out, int out_size, void* d_ws, size_t ws_size,
                              hipStream_t stream) {
    const float* x       = (const float*)d_in[0];
    const float* dist    = (const float*)d_in[1];
    const float* mlp_w1  = (const float*)d_in[2];
    const float* mlp_b1  = (const float*)d_in[3];
    const float* mlp_w2  = (const float*)d_in[4];
    const float* mlp_b2  = (const float*)d_in[5];
    const float* w_stat  = (const float*)d_in[6];
    const float* a_stat  = (const float*)d_in[7];
    const float* W_dis   = (const float*)d_in[8];
    const float* a_dis   = (const float*)d_in[9];
    const float* W_ih    = (const float*)d_in[10];
    const float* W_hh    = (const float*)d_in[11];
    const float* b_ih    = (const float*)d_in[12];
    const float* b_hh    = (const float*)d_in[13];
    const float* lin_w   = (const float*)d_in[14];
    const float* lin_b   = (const float*)d_in[15];
    const float* lin1_w  = (const float*)d_in[16];
    const float* lin1_b  = (const float*)d_in[17];
    const float* lin2_w  = (const float*)d_in[18];
    const float* lin2_b  = (const float*)d_in[19];
    float* out = (float*)d_out;

    char* ws = (char*)d_ws;
    const size_t MB = 1024ull*1024ull;
    _Float16* Ah     = (_Float16*)(ws + 0);                 // 2 MB
    float*    u_     = (float*)   (ws + 2*MB);              // 4 KB
    float*    v_     = (float*)   (ws + 2*MB + 4096);       // 4 KB
    float*    wa_    = (float*)   (ws + 2*MB + 8192);       // 256 B
    _Float16* w1h    = (_Float16*)(ws + 2*MB + 8448);       // 32 KB
    _Float16* w2h    = (_Float16*)(ws + 2*MB + 8448 + 32768);
    _Float16* wihh   = (_Float16*)(ws + 2*MB + 8448 + 65536);     // 128 KB
    _Float16* whhh   = (_Float16*)(ws + 2*MB + 8448 + 65536 + 131072);
    _Float16* linwh  = (_Float16*)(ws + 2*MB + 8448 + 65536 + 262144);
    _Float16* lstmin = (_Float16*)(ws + 4*MB);              // 32 MB
    float*    hsf    = (float*)   (ws + 36*MB);             // 64 MB
    _Float16* hst    = (_Float16*)(ws + 100*MB);            // 32 MB
    float*    sbuf   = (float*)   (ws + 132*MB);            // 512 KB

    prep_kernel<<<4, 256, 0, stream>>>(x, w_stat, a_stat, W_dis, a_dis, u_, v_, wa_);
    convw_kernel<<<704, 256, 0, stream>>>(mlp_w1, mlp_w2, W_ih, W_hh, lin_w,
                                          w1h, w2h, wihh, whhh, linwh);
    attn_kernel<<<NNODE, 256, 0, stream>>>(dist, u_, v_, wa_, Ah);
    gate_kernel<<<NT/16, 256, 0, stream>>>(x, w1h, w2h, mlp_b1, mlp_b2, lstmin);
    lstm_kernel<<<NNODE/16, 256, 0, stream>>>(lstmin, wihh, whhh, b_ih, b_hh, hsf, hst);
    mix_kernel<<<dim3(NNODE/16, TT/2), 256, 0, stream>>>(Ah, hst, hsf, linwh, lin_b,
                                                         lin1_w, lin1_b, sbuf);
    final_kernel<<<128, 256, 0, stream>>>(sbuf, lin2_w, lin2_b, out);
}
